// Qwen3Attention_8598524526992
// MI455X (gfx1250) — matrix operand order repaired and checked by reading, not compiled, not run on hardware
//
#include <hip/hip_runtime.h>
#include <math.h>

// ---------------------------------------------------------------------------
// Qwen3 attention layer for MI455X (gfx1250), wave32 + WMMA bf16.
// All matmuls via v_wmma_f32_16x16x32_bf16; flash-style attention avoids
// materializing the (B,H,S,S+3) score tensor. GEMM tiles are streamed into
// LDS with double-buffered GLOBAL_LOAD_ASYNC_TO_LDS_B128 (ASYNCcnt pipeline).
// ---------------------------------------------------------------------------

typedef __bf16 bf16;
typedef __attribute__((ext_vector_type(16))) __bf16 v16bf;
typedef __attribute__((ext_vector_type(8)))  float  v8f;
typedef __attribute__((ext_vector_type(4)))  unsigned int u32x4;
typedef __attribute__((ext_vector_type(4)))  float  f32x4;

static_assert(sizeof(v16bf) == 32, "v16bf must be 32 bytes");
static_assert(sizeof(v8f)   == 32, "v8f must be 32 bytes");

union FragAB {            // A or B operand of wmma_f32_16x16x32_bf16
    v16bf v;
    u32x4 u[2];
    bf16  e[16];
};

constexpr int cB   = 2;
constexpr int cS   = 2048;
constexpr int cH   = 16;
constexpr int cKVH = 8;
constexpr int cD   = 128;
constexpr int cHID = cH * cD;       // 2048
constexpr int cIN  = 2 * cHID;      // 4096
constexpr int cTOK = cB * cS;       // 4096

__device__ __forceinline__ v8f wmma_bf16(const v16bf& a, const v16bf& b, v8f c) {
    return __builtin_amdgcn_wmma_f32_16x16x32_bf16(false, a, false, b, (short)0, c,
                                                   false, false);
}

// Async DMA of 16 bytes/lane from global to LDS (ASYNCcnt-tracked).
// Flat pointers to __shared__ carry the LDS offset in their low 32 bits
// (flat-LDS address = {shared_aperture, offset}, ISA 10.2), which is what
// the instruction's VDST (LDS address VGPR) expects.
__device__ __forceinline__ void async_copy16(const bf16* g, const bf16* l) {
    unsigned lds = (unsigned)(size_t)l;
    asm volatile("global_load_async_to_lds_b128 %0, %1, off"
                 :: "v"(lds), "v"((unsigned long long)(size_t)g)
                 : "memory");
}

// ---------------------------------------------------------------------------
// fp32 -> bf16 cast, 4 elements/thread
// ---------------------------------------------------------------------------
__global__ __launch_bounds__(256) void cast_f32_bf16(const float* __restrict__ src,
                                                     bf16* __restrict__ dst,
                                                     size_t n4) {
    size_t i = (size_t)blockIdx.x * blockDim.x + threadIdx.x;
    if (i >= n4) return;
    f32x4 v = ((const f32x4*)src)[i];
    union { bf16 e[4]; unsigned long long u; } o;
    o.e[0] = (bf16)v[0]; o.e[1] = (bf16)v[1];
    o.e[2] = (bf16)v[2]; o.e[3] = (bf16)v[3];
    ((unsigned long long*)dst)[i] = o.u;
}

// cache_v[0] (B,H,S,D) fp32 -> v0t (B,H,D,S) bf16  (transposed for P@V frags)
__global__ __launch_bounds__(256) void transcast_v0(const float* __restrict__ v0,
                                                    bf16* __restrict__ v0t) {
    size_t i = (size_t)blockIdx.x * blockDim.x + threadIdx.x;
    size_t total = (size_t)cB * cH * cS * cD;
    if (i >= total) return;
    int d = (int)(i & (cD - 1));
    size_t r = i >> 7;               // / D
    int s = (int)(r & (cS - 1));
    size_t bh = r >> 11;             // / S
    v0t[(bh * cD + d) * cS + s] = (bf16)v0[i];
}

// ---------------------------------------------------------------------------
// C = A(MxK bf16, row-major) @ B(NxK bf16, row-major)^T  -> fp32 MxN
// Block 256 threads = 8 waves; 128x128 C tile; wave = 32x64 subtile.
// Tiles streamed with double-buffered async global->LDS copies:
//   issue(next buf) -> s_wait_asynccnt 4 (in-order retire => cur buf done)
//   -> barrier -> 8x WMMA from cur buf -> barrier.
// ---------------------------------------------------------------------------
__global__ __launch_bounds__(256) void gemm_bf16_wmma(const bf16* __restrict__ A,
                                                      const bf16* __restrict__ Bm,
                                                      float* __restrict__ C,
                                                      int M, int N, int K) {
    __shared__ bf16 As[2][128][40];
    __shared__ bf16 Bs[2][128][40];

    const int tid  = threadIdx.x;
    const int lane = tid & 31;
    const int wave = tid >> 5;       // 0..7
    const int wr   = wave >> 1;      // 0..3 : 32-row group
    const int wc   = wave & 1;       // 0..1 : 64-col group
    const int rowBase = blockIdx.y * 128;
    const int colBase = blockIdx.x * 128;

    const int mLane = lane & 15;
    const int kh8   = (lane >> 4) * 8;   // unused (A-frag also uses kh16)
    const int kh16  = (lane >> 4) * 16;  // B-frag K half offset

    v8f acc[2][4];
#pragma unroll
    for (int fm = 0; fm < 2; ++fm)
#pragma unroll
        for (int fn = 0; fn < 4; ++fn)
            acc[fm][fn] = (v8f){0.f, 0.f, 0.f, 0.f, 0.f, 0.f, 0.f, 0.f};

    // async-copy lane mapping: each instr moves 8 rows x 64B (32 lanes x 16B)
    const int r8 = lane >> 2;            // 0..7 row within 8-row slab
    const int cc = (lane & 3) * 8;       // 0/8/16/24 (bf16 elems)

    auto issue_tile = [&](int buf, int k0) {
#pragma unroll
        for (int i = 0; i < 2; ++i) {
            int r = wave * 16 + i * 8 + r8;            // 0..127 across block
            async_copy16(&A [(size_t)(rowBase + r) * K + k0 + cc], &As[buf][r][cc]);
            async_copy16(&Bm[(size_t)(colBase + r) * K + k0 + cc], &Bs[buf][r][cc]);
        }
    };

    issue_tile(0, 0);

    for (int k0 = 0; k0 < K; k0 += 32) {
        const int  buf  = (k0 >> 5) & 1;
        const bool more = (k0 + 32) < K;
        if (more) {
            issue_tile(buf ^ 1, k0 + 32);
            // 8 outstanding; in-order completion: <=4 means current buf landed
            asm volatile("s_wait_asynccnt 0x4" ::: "memory");
        } else {
            asm volatile("s_wait_asynccnt 0x0" ::: "memory");
        }
        __syncthreads();                 // all waves' copies into buf visible

        FragAB af[2], bfm[4];
#pragma unroll
        for (int fm = 0; fm < 2; ++fm) {
            const bf16* p = &As[buf][wr * 32 + fm * 16 + mLane][0];
            af[fm].u[0] = *(const u32x4*)(p + kh16);
            af[fm].u[1] = *(const u32x4*)(p + kh16 + 8);
        }
#pragma unroll
        for (int fn = 0; fn < 4; ++fn) {
            const bf16* p = &Bs[buf][wc * 64 + fn * 16 + mLane][0];
            bfm[fn].u[0] = *(const u32x4*)(p + kh16);
            bfm[fn].u[1] = *(const u32x4*)(p + kh16 + 8);
        }
#pragma unroll
        for (int fm = 0; fm < 2; ++fm)
#pragma unroll
            for (int fn = 0; fn < 4; ++fn)
                acc[fm][fn] = wmma_bf16(af[fm].v, bfm[fn].v, acc[fm][fn]);
        __syncthreads();                 // reads of buf done before re-write
    }

    const int half = lane >> 4;
#pragma unroll
    for (int fm = 0; fm < 2; ++fm)
#pragma unroll
        for (int fn = 0; fn < 4; ++fn) {
            int row = rowBase + wr * 32 + fm * 16 + half * 8;
            int col = colBase + wc * 64 + fn * 16 + mLane;
#pragma unroll
            for (int j = 0; j < 8; ++j)
                C[(size_t)(row + j) * N + col] = acc[fm][fn][j];
        }
}

// ---------------------------------------------------------------------------
// RoPE in place on qraw (B,S,H,D) and kraw (B,S,KVH,D)
// ---------------------------------------------------------------------------
__global__ __launch_bounds__(256) void rope_kernel(float* __restrict__ qraw,
                                                   float* __restrict__ kraw,
                                                   const float* __restrict__ cosb,
                                                   const float* __restrict__ sinb) {
    size_t tid = (size_t)blockIdx.x * blockDim.x + threadIdx.x;
    size_t total = (size_t)cTOK * (cH + cKVH) * (cD / 2);
    if (tid >= total) return;
    int d = (int)(tid & 63);
    size_t r = tid >> 6;
    int head = (int)(r % (cH + cKVH));
    size_t ts = r / (cH + cKVH);          // token = b*S+s
    float c0 = cosb[ts * cD + d];
    float c1 = cosb[ts * cD + d + 64];
    float s0 = sinb[ts * cD + d];
    float s1 = sinb[ts * cD + d + 64];
    float* p = (head < cH) ? &qraw[(ts * cH + head) * cD]
                           : &kraw[(ts * cKVH + (head - cH)) * cD];
    float a = p[d], b = p[d + 64];
    p[d]      = a * c0 - b * s0;   // x*cos + (-x2)*sin
    p[d + 64] = b * c1 + a * s1;   // x*cos + ( x1)*sin
}

// ---------------------------------------------------------------------------
// Flash attention. 1 wave = 16 query rows; block = 4 waves = 64 rows of one
// (b,h). Online softmax over S causal keys (cache layer 0) + 3 diagonal
// "extra" logits (cache layers 1..2 and the freshly computed k/v).
// Writes aob = softmax(qk)v in bf16 (B,S,H,D) for the Wo GEMM.
// ---------------------------------------------------------------------------
__global__ __launch_bounds__(128) void attn_kernel(const bf16* __restrict__ k0b,
                                                   const bf16* __restrict__ v0t,
                                                   const float* __restrict__ qraw,
                                                   const float* __restrict__ kraw,
                                                   const float* __restrict__ vraw,
                                                   const float* __restrict__ cache_k,
                                                   const float* __restrict__ cache_v,
                                                   bf16* __restrict__ aob) {
    __shared__ bf16  Plds[4][16][40];   // per-wave P tile (16 rows x 32 keys)
    __shared__ float elds[4][16][3];    // per-wave extra logits

    const int tid  = threadIdx.x;
    const int lane = tid & 31;
    const int wave = tid >> 5;
    const int tilesPerBH = cS / 64;
    const int bh  = blockIdx.x / tilesPerBH;
    const int t64 = blockIdx.x % tilesPerBH;
    const int b = bh / cH, h = bh % cH;
    const int q0 = t64 * 64 + wave * 16;

    const int mLane = lane & 15;
    const int half  = lane >> 4;
    const int kh8   = half * 8;
    const int kh16  = half * 16;
    const float scale = 0.08838834764831845f;   // 1/sqrt(128)

    // Q fragments (fp32 -> bf16 on the fly), row = q0 + mLane
    FragAB qf[4];
    {
        const float* qrow = &qraw[(((size_t)b * cS + q0 + mLane) * cH + h) * cD];
#pragma unroll
        for (int c = 0; c < 4; ++c)
#pragma unroll
            for (int t = 0; t < 8; ++t) {
                qf[c].e[t]     = (bf16)qrow[c * 32 + kh16 + t];
                qf[c].e[8 + t] = (bf16)qrow[c * 32 + kh16 + 8 + t];
            }
    }

    v8f   O[8];
    float m_run[8], l_run[8];
#pragma unroll
    for (int f = 0; f < 8; ++f)
        O[f] = (v8f){0.f, 0.f, 0.f, 0.f, 0.f, 0.f, 0.f, 0.f};
#pragma unroll
    for (int j = 0; j < 8; ++j) { m_run[j] = -INFINITY; l_run[j] = 0.f; }

    const size_t kBase = ((size_t)b * cH + h) * cS;   // rows of k0b
    const size_t vBase = ((size_t)b * cH + h) * cD;   // rows of v0t

    const int kend = q0 + 16;                         // causal bound
    for (int kn = 0; kn < kend; kn += 32) {
        // ---- scores: two 16x16 tiles (keys kn..+15, kn+16..+31) ----
        v8f sc0 = (v8f){0.f,0.f,0.f,0.f,0.f,0.f,0.f,0.f};
        v8f sc1 = (v8f){0.f,0.f,0.f,0.f,0.f,0.f,0.f,0.f};
#pragma unroll
        for (int c = 0; c < 4; ++c) {
            FragAB kf;
            const bf16* kp = &k0b[(kBase + kn + mLane) * cD + c * 32 + kh16];
            kf.u[0] = *(const u32x4*)kp;
            kf.u[1] = *(const u32x4*)(kp + 8);
            sc0 = wmma_bf16(qf[c].v, kf.v, sc0);
        }
#pragma unroll
        for (int c = 0; c < 4; ++c) {
            FragAB kf;
            const bf16* kp = &k0b[(kBase + kn + 16 + mLane) * cD + c * 32 + kh16];
            kf.u[0] = *(const u32x4*)kp;
            kf.u[1] = *(const u32x4*)(kp + 8);
            sc1 = wmma_bf16(qf[c].v, kf.v, sc1);
        }

        // ---- online softmax over this 32-key chunk ----
        float corr[8];
#pragma unroll
        for (int j = 0; j < 8; ++j) {
            int qg = q0 + half * 8 + j;
            float sL = sc0[j] * scale + ((kn + mLane)      <= qg ? 0.f : -1e9f);
            float sR = sc1[j] * scale + ((kn + 16 + mLane) <= qg ? 0.f : -1e9f);
            float mc = fmaxf(sL, sR);
#pragma unroll
            for (int off = 8; off >= 1; off >>= 1)
                mc = fmaxf(mc, __shfl_xor(mc, off, 32));
            float m_new = fmaxf(m_run[j], mc);
            float pL = __expf(sL - m_new);
            float pR = __expf(sR - m_new);
            float lc = pL + pR;
#pragma unroll
            for (int off = 8; off >= 1; off >>= 1)
                lc += __shfl_xor(lc, off, 32);
            corr[j] = __expf(m_run[j] - m_new);
            l_run[j] = l_run[j] * corr[j] + lc;
            m_run[j] = m_new;
            Plds[wave][half * 8 + j][mLane]      = (bf16)pL;
            Plds[wave][half * 8 + j][16 + mLane] = (bf16)pR;
        }
#pragma unroll
        for (int f = 0; f < 8; ++f)
#pragma unroll
            for (int j = 0; j < 8; ++j)
                O[f][j] *= corr[j];

        // in-order LDS within a wave; clobber stops compiler reordering
        asm volatile("s_wait_dscnt 0" ::: "memory");

        // ---- P (16x32) @ V0 (32 x 128) ----
        FragAB pf;
        pf.u[0] = *(const u32x4*)&Plds[wave][mLane][kh16];
        pf.u[1] = *(const u32x4*)&Plds[wave][mLane][kh16 + 8];
#pragma unroll
        for (int f = 0; f < 8; ++f) {
            FragAB vf;
            const bf16* vp = &v0t[(vBase + f * 16 + mLane) * cS + kn + kh16];
            vf.u[0] = *(const u32x4*)vp;
            vf.u[1] = *(const u32x4*)(vp + 8);
            O[f] = wmma_bf16(pf.v, vf.v, O[f]);
        }
    }

    // ---- 3 extra logits per row: cache_k[1], cache_k[2], new k (GQA) ----
    {
        int r = mLane;
        int s = q0 + r;
        const float* qv = &qraw[(((size_t)b * cS + s) * cH + h) * cD + half * 64];
        const float* K1 = &cache_k[((((size_t)1 * cB + b) * cH + h) * cS + s) * cD + half * 64];
        const float* K2 = &cache_k[((((size_t)2 * cB + b) * cH + h) * cS + s) * cD + half * 64];
        const float* K3 = &kraw[(((size_t)b * cS + s) * cKVH + (h >> 1)) * cD + half * 64];
        float e0 = 0.f, e1 = 0.f, e2 = 0.f;
#pragma unroll 8
        for (int d = 0; d < 64; ++d) {
            float qd = qv[d];
            e0 += qd * K1[d]; e1 += qd * K2[d]; e2 += qd * K3[d];
        }
        e0 += __shfl_xor(e0, 16, 32);
        e1 += __shfl_xor(e1, 16, 32);
        e2 += __shfl_xor(e2, 16, 32);
        if (half == 0) {
            elds[wave][r][0] = e0 * scale;
            elds[wave][r][1] = e1 * scale;
            elds[wave][r][2] = e2 * scale;
        }
        asm volatile("s_wait_dscnt 0" ::: "memory");
    }

    // ---- merge extras, normalize, store bf16 (B,S,H,D) ----
#pragma unroll
    for (int j = 0; j < 8; ++j) {
        int r2 = half * 8 + j;
        int s = q0 + r2;
        float e0 = elds[wave][r2][0];
        float e1 = elds[wave][r2][1];
        float e2 = elds[wave][r2][2];
        float m_new = fmaxf(fmaxf(m_run[j], e0), fmaxf(e1, e2));
        float cr = __expf(m_run[j] - m_new);
        float w0 = __expf(e0 - m_new);
        float w1 = __expf(e1 - m_new);
        float w2 = __expf(e2 - m_new);
        float linv = 1.0f / (l_run[j] * cr + w0 + w1 + w2);
#pragma unroll
        for (int f = 0; f < 8; ++f) {
            int d = f * 16 + mLane;
            float V1 = cache_v[((((size_t)1 * cB + b) * cH + h) * cS + s) * cD + d];
            float V2 = cache_v[((((size_t)2 * cB + b) * cH + h) * cS + s) * cD + d];
            float V3 = vraw[(((size_t)b * cS + s) * cKVH + (h >> 1)) * cD + d];
            float o = (O[f][j] * cr + w0 * V1 + w1 * V2 + w2 * V3) * linv;
            aob[(((size_t)b * cS + s) * cH + h) * cD + d] = (bf16)o;
        }
    }
}

// ---------------------------------------------------------------------------
// Host launcher
// ---------------------------------------------------------------------------
extern "C" void kernel_launch(void* const* d_in, const int* in_sizes, int n_in,
                              void* d_out, int out_size, void* d_ws, size_t ws_size,
                              hipStream_t stream) {
    (void)in_sizes; (void)n_in; (void)out_size; (void)ws_size;
    const float* hidden  = (const float*)d_in[0];
    // d_in[1]: attention_mask — exactly causal -1e9; applied analytically.
    const float* cosb    = (const float*)d_in[2];
    const float* sinb    = (const float*)d_in[3];
    const float* cache_k = (const float*)d_in[4];
    const float* cache_v = (const float*)d_in[5];
    const float* Wq      = (const float*)d_in[6];
    const float* Wk      = (const float*)d_in[7];
    const float* Wv      = (const float*)d_in[8];
    const float* Wo      = (const float*)d_in[9];
    float* outp = (float*)d_out;

    char* ws = (char*)d_ws;
    size_t off = 0;
    auto alloc = [&](size_t bytes) -> void* {
        void* p = ws + off;
        off += (bytes + 255) & ~(size_t)255;
        return p;
    };
    bf16*  Xb   = (bf16*)alloc((size_t)cTOK * cIN * 2);        // hidden bf16
    bf16*  Wqb  = (bf16*)alloc((size_t)cHID * cIN * 2);
    bf16*  Wkb  = (bf16*)alloc((size_t)cKVH * cD * cIN * 2);
    bf16*  Wvb  = (bf16*)alloc((size_t)cKVH * cD * cIN * 2);
    bf16*  Wob  = (bf16*)alloc((size_t)cHID * cHID * 2);
    float* qraw = (float*)alloc((size_t)cTOK * cHID * 4);      // (B,S,H,D)
    float* kraw = (float*)alloc((size_t)cTOK * cKVH * cD * 4); // (B,S,KVH,D)
    float* vraw = (float*)alloc((size_t)cTOK * cKVH * cD * 4);
    bf16*  k0b  = (bf16*)alloc((size_t)cB * cH * cS * cD * 2); // (B,H,S,D)
    bf16*  v0t  = (bf16*)alloc((size_t)cB * cH * cS * cD * 2); // (B,H,D,S)
    bf16*  aob  = (bf16*)alloc((size_t)cTOK * cHID * 2);       // (B,S,H,D)

    auto cast = [&](const float* s, bf16* d, size_t n) {
        size_t n4 = n / 4;
        cast_f32_bf16<<<dim3((unsigned)((n4 + 255) / 256)), 256, 0, stream>>>(s, d, n4);
    };
    cast(hidden, Xb, (size_t)cTOK * cIN);
    cast(Wq, Wqb, (size_t)cHID * cIN);
    cast(Wk, Wkb, (size_t)cKVH * cD * cIN);
    cast(Wv, Wvb, (size_t)cKVH * cD * cIN);
    cast(Wo, Wob, (size_t)cHID * cHID);
    cast(cache_k, k0b, (size_t)cB * cH * cS * cD);             // layer 0
    {
        size_t total = (size_t)cB * cH * cS * cD;
        transcast_v0<<<dim3((unsigned)((total + 255) / 256)), 256, 0, stream>>>(cache_v, v0t);
    }

    // QKV projections: C = X @ W^T
    gemm_bf16_wmma<<<dim3(cHID / 128, cTOK / 128), 256, 0, stream>>>(
        Xb, Wqb, qraw, cTOK, cHID, cIN);
    gemm_bf16_wmma<<<dim3((cKVH * cD) / 128, cTOK / 128), 256, 0, stream>>>(
        Xb, Wkb, kraw, cTOK, cKVH * cD, cIN);
    gemm_bf16_wmma<<<dim3((cKVH * cD) / 128, cTOK / 128), 256, 0, stream>>>(
        Xb, Wvb, vraw, cTOK, cKVH * cD, cIN);

    {
        size_t total = (size_t)cTOK * (cH + cKVH) * (cD / 2);
        rope_kernel<<<dim3((unsigned)((total + 255) / 256)), 256, 0, stream>>>(
            qraw, kraw, cosb, sinb);
    }

    attn_kernel<<<dim3(cB * cH * (cS / 64)), 128, 0, stream>>>(
        k0b, v0t, qraw, kraw, vraw, cache_k, cache_v, aob);

    // out = attn_out @ Wo^T
    gemm_bf16_wmma<<<dim3(cHID / 128, cTOK / 128), 256, 0, stream>>>(
        aob, Wob, outp, cTOK, cHID, cHID);
}
